// PFGRUCell_71631464563003
// MI455X (gfx1250) — matrix operand order repaired and checked by reading, not compiled, not run on hardware
//
#include <hip/hip_runtime.h>
#include <hip/hip_bf16.h>
#include <math.h>

typedef __attribute__((ext_vector_type(16))) _Float16 v16h;
typedef __attribute__((ext_vector_type(8)))  _Float16 v8h;
typedef __attribute__((ext_vector_type(8)))  float    v8f;

#define K_PART 64
#define B_SZ   4096
#define H_SZ   256
#define IN_SZ  128
#define HXW    384                 // H + IN
#define NROWS  (K_PART * B_SZ)     // 262144
#define ALPHA_ 0.5f
#define BN_EPS_ 1e-5f

__device__ __forceinline__ float sigmoidf(float x) { return 1.0f / (1.0f + __expf(-x)); }
__device__ __forceinline__ float softplusf(float x) {
    if (x > 20.0f)  return x;
    if (x < -20.0f) return __expf(x);
    return log1pf(__expf(x));
}
__device__ __forceinline__ v16h combine8(v8h lo, v8h hi) {
    return __builtin_shufflevector(lo, hi, 0,1,2,3,4,5,6,7,8,9,10,11,12,13,14,15);
}

// CDNA5 async global->LDS copy, 16 bytes per lane (ISA 15.18.3 op 98).
// vdst = LDS byte address (low 32 bits of generic shared pointer), vaddr = 64-bit
// global address. Tracked by ASYNCcnt; completion via s_wait_asynccnt.
__device__ __forceinline__ void async_copy16(unsigned lds_addr, unsigned long long gaddr) {
    asm volatile("global_load_async_to_lds_b128 %0, %1, off"
                 :: "v"(lds_addr), "v"(gaddr) : "memory");
}
__device__ __forceinline__ void wait_async0() {
    asm volatile("s_wait_asynccnt 0" ::: "memory");
}

// ---------------------------------------------------------------------------
// k0: zero BN accumulators (ws is poisoned before timing; re-zero every call)
// ---------------------------------------------------------------------------
__global__ void k0_zero(float* ksum, float* ksumsq) {
    int i = threadIdx.x;
    if (i < K_PART) { ksum[i] = 0.0f; ksumsq[i] = 0.0f; }
}

// ---------------------------------------------------------------------------
// k0b: pack f32 weights -> f16, pre-swizzled per 32-row k-chunk, col-major
//      [kk][col*32 + k%32] so LDS staging is a contiguous async byte copy.
// ---------------------------------------------------------------------------
__global__ void k0b_pack(const float* __restrict__ W, _Float16* __restrict__ out,
                         int ncols, int nrows) {
    int i = blockIdx.x * blockDim.x + threadIdx.x;
    if (i >= nrows * ncols) return;
    int k = i / ncols, col = i % ncols;
    int kk = k >> 5, kr = k & 31;
    out[(size_t)kk * ((size_t)ncols * 32) + (size_t)col * 32 + kr] = (_Float16)W[i];
}

// ---------------------------------------------------------------------------
// k1: fused WMMA GEMMs (z, r, n) on a 32-row stripe per 256-thread block.
//     Weights arrive via global_load_async_to_lds_b128 from the packed copies.
// ---------------------------------------------------------------------------
__global__ __launch_bounds__(256)
void k1_gemm_zrn(const float* __restrict__ input, const float* __restrict__ h0,
                 const float* __restrict__ eps,
                 const _Float16* __restrict__ Wzt, const _Float16* __restrict__ Wrt,
                 const _Float16* __restrict__ Wnt,
                 const float* __restrict__ bz, const float* __restrict__ br,
                 const float* __restrict__ bn_b,
                 float* __restrict__ nraw, _Float16* __restrict__ zbuf,
                 float* __restrict__ ksum, float* __restrict__ ksumsq)
{
    __shared__ __align__(16) _Float16 s_hx[32 * HXW];   // 24 KB
    __shared__ __align__(16) _Float16 s_a2[32 * HXW];   // 24 KB
    __shared__ __align__(16) _Float16 s_B[16384];       // 32 KB weight stage
    __shared__ float s_red[16];

    const int tid  = threadIdx.x;
    const int lane = tid & 31;
    const int w    = tid >> 5;
    const int r0   = blockIdx.x * 32;
    const int kpart = r0 / B_SZ;
    const unsigned ldsB = (unsigned)(uintptr_t)&s_B[0];

    // ---- stage hx = [h0 | input] as f16 ----
    for (int i = tid; i < 32 * HXW; i += 256) {
        int row = i / HXW, col = i % HXW;
        float v = (col < H_SZ) ? h0[(size_t)(r0 + row) * H_SZ + col]
                               : input[(size_t)(r0 + row) * IN_SZ + (col - H_SZ)];
        s_hx[i] = (_Float16)v;
    }

    // ---- z / r GEMMs: wave w owns col tiles {2w,2w+1}, row tiles {0,1} ----
    v8f az[2][2] = {{{}, {}}, {{}, {}}};
    v8f ar[2][2] = {{{}, {}}, {{}, {}}};
    const int nt0 = w * 2;
    for (int kk = 0; kk < 12; ++kk) {
        const int k0 = kk * 32;
        __syncthreads();
        {
            unsigned long long gz = (unsigned long long)(uintptr_t)Wzt + (size_t)kk * 16384;
            unsigned long long gr = (unsigned long long)(uintptr_t)Wrt + (size_t)kk * 16384;
            #pragma unroll
            for (int it = 0; it < 4; ++it) {
                unsigned off = (unsigned)(tid * 16 + it * 4096);
                async_copy16(ldsB + off,         gz + off);
                async_copy16(ldsB + 16384 + off, gr + off);
            }
            wait_async0();
        }
        __syncthreads();
        const int mm = lane & 15, hf = lane >> 4;
        const int kh = hf * 8;
        const int n0 = (nt0    ) * 16 + mm;
        const int n1 = (nt0 + 1) * 16 + mm;
        v16h a0, a1;
        {
            const _Float16* ap0 = &s_hx[(0 * 16 + mm) * HXW + k0 + hf * 8];
            const _Float16* ap1 = &s_hx[(1 * 16 + mm) * HXW + k0 + hf * 8];
            a0 = combine8(*(const v8h*)ap0, *(const v8h*)(ap0 + 16));
            a1 = combine8(*(const v8h*)ap1, *(const v8h*)(ap1 + 16));
        }
        v16h bz0f = combine8(*(const v8h*)&s_B[n0 * 32 + kh], *(const v8h*)&s_B[n0 * 32 + kh + 16]);
        v16h bz1f = combine8(*(const v8h*)&s_B[n1 * 32 + kh], *(const v8h*)&s_B[n1 * 32 + kh + 16]);
        v16h br0f = combine8(*(const v8h*)&s_B[8192 + n0 * 32 + kh], *(const v8h*)&s_B[8192 + n0 * 32 + kh + 16]);
        v16h br1f = combine8(*(const v8h*)&s_B[8192 + n1 * 32 + kh], *(const v8h*)&s_B[8192 + n1 * 32 + kh + 16]);
        az[0][0] = __builtin_amdgcn_wmma_f32_16x16x32_f16(false, a0, false, bz0f, (short)0, az[0][0], false, false);
        az[1][0] = __builtin_amdgcn_wmma_f32_16x16x32_f16(false, a1, false, bz0f, (short)0, az[1][0], false, false);
        az[0][1] = __builtin_amdgcn_wmma_f32_16x16x32_f16(false, a0, false, bz1f, (short)0, az[0][1], false, false);
        az[1][1] = __builtin_amdgcn_wmma_f32_16x16x32_f16(false, a1, false, bz1f, (short)0, az[1][1], false, false);
        ar[0][0] = __builtin_amdgcn_wmma_f32_16x16x32_f16(false, a0, false, br0f, (short)0, ar[0][0], false, false);
        ar[1][0] = __builtin_amdgcn_wmma_f32_16x16x32_f16(false, a1, false, br0f, (short)0, ar[1][0], false, false);
        ar[0][1] = __builtin_amdgcn_wmma_f32_16x16x32_f16(false, a0, false, br1f, (short)0, ar[0][1], false, false);
        ar[1][1] = __builtin_amdgcn_wmma_f32_16x16x32_f16(false, a1, false, br1f, (short)0, ar[1][1], false, false);
    }
    __syncthreads();
    // activations: z -> global f16, r*h0 -> s_a2
    {
        const int mm = lane & 15;
        const int nbase = nt0 * 16 + mm;
        const int mof = (lane >> 4) * 8;
        #pragma unroll
        for (int mt = 0; mt < 2; ++mt) {
            #pragma unroll
            for (int j = 0; j < 8; ++j) {
                int m = mt * 16 + j + mof;
                {
                    int col = nbase;
                    float z = sigmoidf(az[mt][0][j] + bz[col]);
                    zbuf[(size_t)(r0 + m) * H_SZ + col] = (_Float16)z;
                    float r = sigmoidf(ar[mt][0][j] + br[col]);
                    s_a2[m * HXW + col] = (_Float16)(r * (float)s_hx[m * HXW + col]);
                }
                {
                    int col = nbase + 16;
                    float z = sigmoidf(az[mt][1][j] + bz[col]);
                    zbuf[(size_t)(r0 + m) * H_SZ + col] = (_Float16)z;
                    float r = sigmoidf(ar[mt][1][j] + br[col]);
                    s_a2[m * HXW + col] = (_Float16)(r * (float)s_hx[m * HXW + col]);
                }
            }
        }
    }
    for (int i = tid; i < 32 * IN_SZ; i += 256) {
        int row = i / IN_SZ, col = i % IN_SZ;
        s_a2[row * HXW + H_SZ + col] = s_hx[row * HXW + H_SZ + col];
    }

    // ---- n GEMM: 32 col tiles, wave w owns {4w..4w+3}, row tiles {0,1} ----
    v8f an[2][4] = {{{}, {}, {}, {}}, {{}, {}, {}, {}}};
    for (int kk = 0; kk < 12; ++kk) {
        const int k0 = kk * 32;
        __syncthreads();
        {
            unsigned long long gn = (unsigned long long)(uintptr_t)Wnt + (size_t)kk * 32768;
            #pragma unroll
            for (int it = 0; it < 8; ++it) {
                unsigned off = (unsigned)(tid * 16 + it * 4096);
                async_copy16(ldsB + off, gn + off);
            }
            wait_async0();
        }
        __syncthreads();
        const int mm = lane & 15, hf = lane >> 4;
        const int kh = hf * 8;
        v16h a0, a1;
        {
            const _Float16* ap0 = &s_a2[(0 * 16 + mm) * HXW + k0 + hf * 8];
            const _Float16* ap1 = &s_a2[(1 * 16 + mm) * HXW + k0 + hf * 8];
            a0 = combine8(*(const v8h*)ap0, *(const v8h*)(ap0 + 16));
            a1 = combine8(*(const v8h*)ap1, *(const v8h*)(ap1 + 16));
        }
        #pragma unroll
        for (int t = 0; t < 4; ++t) {
            int n = (w * 4 + t) * 16 + mm;
            v16h bf = combine8(*(const v8h*)&s_B[n * 32 + kh], *(const v8h*)&s_B[n * 32 + kh + 16]);
            an[0][t] = __builtin_amdgcn_wmma_f32_16x16x32_f16(false, a0, false, bf, (short)0, an[0][t], false, false);
            an[1][t] = __builtin_amdgcn_wmma_f32_16x16x32_f16(false, a1, false, bf, (short)0, an[1][t], false, false);
        }
    }
    // mu/var exchange through LDS (alias staging buffer; 16 rows per pass)
    float* s_nbuf = reinterpret_cast<float*>(s_B);
    float s = 0.0f, s2 = 0.0f;
    #pragma unroll
    for (int half = 0; half < 2; ++half) {
        __syncthreads();
        {
            const int mm = lane & 15;
            const int mof = (lane >> 4) * 8;
            #pragma unroll
            for (int t = 0; t < 4; ++t) {
                int col = (w * 4 + t) * 16 + mm;
                #pragma unroll
                for (int j = 0; j < 8; ++j)
                    s_nbuf[(j + mof) * 512 + col] = an[half][t][j] + bn_b[col];
            }
        }
        __syncthreads();
        for (int i = tid; i < 16 * H_SZ; i += 256) {
            int row = i / H_SZ, col = i % H_SZ;
            int grow = r0 + half * 16 + row;
            float mu = s_nbuf[row * 512 + col];
            float va = s_nbuf[row * 512 + H_SZ + col];
            float e  = eps[(size_t)grow * H_SZ + col];
            float nr = mu + e * softplusf(va);
            nraw[(size_t)grow * H_SZ + col] = nr;
            s += nr; s2 += nr * nr;
        }
    }
    for (int off = 16; off > 0; off >>= 1) {
        s  += __shfl_xor(s,  off, 32);
        s2 += __shfl_xor(s2, off, 32);
    }
    if (lane == 0) { s_red[w] = s; s_red[8 + w] = s2; }
    __syncthreads();
    if (tid == 0) {
        float ts = 0.0f, t2 = 0.0f;
        for (int i = 0; i < 8; ++i) { ts += s_red[i]; t2 += s_red[8 + i]; }
        atomicAdd(&ksum[kpart], ts);
        atomicAdd(&ksumsq[kpart], t2);
    }
}

// ---------------------------------------------------------------------------
// k2: finalize per-k BN stats
// ---------------------------------------------------------------------------
__global__ void k2_stats(const float* ksum, const float* ksumsq, float* mean, float* rstd) {
    int k = threadIdx.x;
    if (k < K_PART) {
        float inv = 1.0f / (float)(B_SZ * H_SZ);
        float m = ksum[k] * inv;
        float v = ksumsq[k] * inv - m * m;
        mean[k] = m;
        rstd[k] = rsqrtf(v + BN_EPS_);
    }
}

// ---------------------------------------------------------------------------
// k3: BN + leaky ReLU + h1 = (1-z)*n + z*h0 (in place over nraw) + Wobs GEMV
// ---------------------------------------------------------------------------
__global__ __launch_bounds__(256)
void k3_bn_h1_obs(const float* __restrict__ input, const float* __restrict__ h0,
                  const float* __restrict__ p0, const _Float16* __restrict__ zbuf,
                  const float* __restrict__ Wobs, const float* __restrict__ bobs,
                  const float* __restrict__ gamma, const float* __restrict__ beta,
                  const float* __restrict__ mean, const float* __restrict__ rstd,
                  float* __restrict__ h1, float* __restrict__ p1raw)
{
    const int tid = threadIdx.x, lane = tid & 31, w = tid >> 5;
    const int r0 = blockIdx.x * 16;
    for (int rr = 0; rr < 2; ++rr) {
        int row = r0 + w * 2 + rr;
        int k = row / B_SZ;
        float mn = mean[k], rs = rstd[k], g = gamma[k], bb = beta[k];
        float dot = 0.0f;
        for (int j = 0; j < 8; ++j) {
            int col = lane * 8 + j;
            float nr = h1[(size_t)row * H_SZ + col];
            float nn = (nr - mn) * rs * g + bb;
            nn = (nn >= 0.0f) ? nn : 0.01f * nn;
            float z = (float)zbuf[(size_t)row * H_SZ + col];
            float hh = (1.0f - z) * nn + z * h0[(size_t)row * H_SZ + col];
            h1[(size_t)row * H_SZ + col] = hh;
            dot += hh * Wobs[col];
        }
        for (int j = 0; j < 4; ++j) {
            int col = lane * 4 + j;
            dot += input[(size_t)row * IN_SZ + col] * Wobs[H_SZ + col];
        }
        for (int off = 16; off > 0; off >>= 1) dot += __shfl_xor(dot, off, 32);
        if (lane == 0) p1raw[row] = dot + bobs[0] + p0[row];
    }
}

// ---------------------------------------------------------------------------
// k4: per-b logsumexp over K; normalize p1 in place; build resampling CDF
// ---------------------------------------------------------------------------
__global__ void k4_lse_cdf(float* __restrict__ p1, float* __restrict__ cdf) {
    int b = blockIdx.x * blockDim.x + threadIdx.x;
    if (b >= B_SZ) return;
    float mx = -1e30f;
    for (int k = 0; k < K_PART; ++k) mx = fmaxf(mx, p1[(size_t)k * B_SZ + b]);
    float sum = 0.0f;
    for (int k = 0; k < K_PART; ++k) sum += __expf(p1[(size_t)k * B_SZ + b] - mx);
    float lse = mx + __logf(sum);
    float cum = 0.0f;
    for (int k = 0; k < K_PART; ++k) {
        float pn = p1[(size_t)k * B_SZ + b] - lse;
        p1[(size_t)k * B_SZ + b] = pn;
        cum += ALPHA_ * __expf(pn) + (1.0f - ALPHA_) / (float)K_PART;
        cdf[(size_t)k * B_SZ + b] = cum;
    }
}

// ---------------------------------------------------------------------------
// k5: categorical draw per (k,b) via deterministic hash; compute log weights
// ---------------------------------------------------------------------------
__device__ __forceinline__ unsigned hash_u32(unsigned x) {
    x ^= 42u * 0x9E3779B9u;
    x *= 0x85EBCA6Bu; x ^= x >> 13; x *= 0xC2B2AE35u; x ^= x >> 16;
    return x;
}
__global__ void k5_sample(const float* __restrict__ p1n, const float* __restrict__ cdf,
                          int* __restrict__ flatidx, float* __restrict__ pnraw) {
    int i = blockIdx.x * blockDim.x + threadIdx.x;
    if (i >= NROWS) return;
    int b = i % B_SZ;
    unsigned h = hash_u32((unsigned)i);
    float total = cdf[(size_t)(K_PART - 1) * B_SZ + b];
    float u = ((h >> 8) * (1.0f / 16777216.0f)) * total;
    int idx = K_PART - 1;
    for (int k = 0; k < K_PART; ++k) {
        if (cdf[(size_t)k * B_SZ + b] >= u) { idx = k; break; }
    }
    int fl = idx * B_SZ + b;
    flatidx[i] = fl;
    float e = __expf(p1n[fl]);
    float pw = e / (ALPHA_ * e + (1.0f - ALPHA_) / (float)K_PART);
    pnraw[i] = __logf(pw);
}

// ---------------------------------------------------------------------------
// k6: renormalize pn over K, write output tail
// ---------------------------------------------------------------------------
__global__ void k6_pn(const float* __restrict__ pnraw, float* __restrict__ out_pn) {
    int b = blockIdx.x * blockDim.x + threadIdx.x;
    if (b >= B_SZ) return;
    float mx = -1e30f;
    for (int k = 0; k < K_PART; ++k) mx = fmaxf(mx, pnraw[(size_t)k * B_SZ + b]);
    float sum = 0.0f;
    for (int k = 0; k < K_PART; ++k) sum += __expf(pnraw[(size_t)k * B_SZ + b] - mx);
    float lse = mx + __logf(sum);
    for (int k = 0; k < K_PART; ++k)
        out_pn[(size_t)k * B_SZ + b] = pnraw[(size_t)k * B_SZ + b] - lse;
}

// ---------------------------------------------------------------------------
// k7: gather resampled h rows into output head (float4-coalesced)
// ---------------------------------------------------------------------------
__global__ void k7_gather(const float* __restrict__ h1, const int* __restrict__ flatidx,
                          float* __restrict__ out_h) {
    int row = blockIdx.x;
    int src = flatidx[row];
    int t = threadIdx.x;  // 64 threads, 1 float4 each
    const float4* s = (const float4*)(h1 + (size_t)src * H_SZ);
    float4*       d = (float4*)(out_h + (size_t)row * H_SZ);
    d[t] = s[t];
}

// ---------------------------------------------------------------------------
extern "C" void kernel_launch(void* const* d_in, const int* in_sizes, int n_in,
                              void* d_out, int out_size, void* d_ws, size_t ws_size,
                              hipStream_t stream) {
    (void)in_sizes; (void)n_in; (void)out_size; (void)ws_size;
    const float* input = (const float*)d_in[0];
    const float* h0    = (const float*)d_in[1];
    const float* p0    = (const float*)d_in[2];
    const float* eps   = (const float*)d_in[3];
    const float* Wz    = (const float*)d_in[4];
    const float* bz    = (const float*)d_in[5];
    const float* Wr    = (const float*)d_in[6];
    const float* br    = (const float*)d_in[7];
    const float* Wn    = (const float*)d_in[8];
    const float* bn_b  = (const float*)d_in[9];
    const float* Wobs  = (const float*)d_in[10];
    const float* bobs  = (const float*)d_in[11];
    const float* gamma = (const float*)d_in[12];
    const float* beta  = (const float*)d_in[13];

    char* ws = (char*)d_ws;
    size_t off = 0;
    float*    buf1   = (float*)(ws + off);    off += (size_t)NROWS * H_SZ * 4; // nraw -> h1
    _Float16* zbuf   = (_Float16*)(ws + off); off += (size_t)NROWS * H_SZ * 2;
    float*    p1raw  = (float*)(ws + off);    off += (size_t)NROWS * 4;
    float*    cdf    = (float*)(ws + off);    off += (size_t)NROWS * 4;
    float*    pnraw  = (float*)(ws + off);    off += (size_t)NROWS * 4;
    int*      flati  = (int*)(ws + off);      off += (size_t)NROWS * 4;
    _Float16* wzt    = (_Float16*)(ws + off); off += (size_t)HXW * H_SZ * 2;
    _Float16* wrt    = (_Float16*)(ws + off); off += (size_t)HXW * H_SZ * 2;
    _Float16* wnt    = (_Float16*)(ws + off); off += (size_t)HXW * 512 * 2;
    float*    ksum   = (float*)(ws + off);    off += K_PART * 4;
    float*    ksumsq = (float*)(ws + off);    off += K_PART * 4;
    float*    meanv  = (float*)(ws + off);    off += K_PART * 4;
    float*    rstdv  = (float*)(ws + off);    off += K_PART * 4;

    float* out_h  = (float*)d_out;
    float* out_pn = out_h + (size_t)NROWS * H_SZ;

    k0_zero<<<1, 64, 0, stream>>>(ksum, ksumsq);
    k0b_pack<<<(HXW * H_SZ) / 256, 256, 0, stream>>>(Wz, wzt, H_SZ, HXW);
    k0b_pack<<<(HXW * H_SZ) / 256, 256, 0, stream>>>(Wr, wrt, H_SZ, HXW);
    k0b_pack<<<(HXW * 512) / 256, 256, 0, stream>>>(Wn, wnt, 512, HXW);
    k1_gemm_zrn<<<NROWS / 32, 256, 0, stream>>>(input, h0, eps, wzt, wrt, wnt,
                                                bz, br, bn_b, buf1, zbuf, ksum, ksumsq);
    k2_stats<<<1, 64, 0, stream>>>(ksum, ksumsq, meanv, rstdv);
    k3_bn_h1_obs<<<NROWS / 16, 256, 0, stream>>>(input, h0, p0, zbuf, Wobs, bobs,
                                                 gamma, beta, meanv, rstdv, buf1, p1raw);
    k4_lse_cdf<<<B_SZ / 256, 256, 0, stream>>>(p1raw, cdf);
    k5_sample<<<NROWS / 256, 256, 0, stream>>>(p1raw, cdf, flati, pnraw);
    k6_pn<<<B_SZ / 256, 256, 0, stream>>>(pnraw, out_pn);
    k7_gather<<<NROWS, 64, 0, stream>>>(buf1, flati, out_h);
}
